// AttentionSubsample_9801115369628
// MI455X (gfx1250) — matrix operand order repaired and checked by reading, not compiled, not run on hardware
//
#include <hip/hip_runtime.h>
#include <hip/hip_bf16.h>

// ---------- CDNA5 WMMA types ----------
typedef __attribute__((ext_vector_type(16))) __bf16 v16bf;
typedef __attribute__((ext_vector_type(8)))  __bf16 v8bf;
typedef __attribute__((ext_vector_type(8)))  float  v8f;
typedef __attribute__((ext_vector_type(4)))  unsigned int v4u;
typedef __attribute__((ext_vector_type(8)))  unsigned int v8u;

#define BATCH   64
#define CIN     256
#define NHEAD   8
#define KDIM    16
#define DVDIM   32
#define NKV     784   // 28*28
#define NQ      196   // 14*14
#define ATTN_LD 800   // NKV padded to multiple of 32

__device__ __forceinline__ float hswish(float x) {
  return x * fminf(fmaxf(x + 3.f, 0.f), 6.f) * (1.f / 6.f);
}

__device__ __forceinline__ v8bf zero8() {
  v8bf z;
#pragma unroll
  for (int i = 0; i < 8; ++i) z[i] = (__bf16)0.f;
  return z;
}

__device__ __forceinline__ v8f wmma_bf16(v16bf a, v16bf b, v8f c) {
  // D = A(16x32 bf16) * B(32x16 bf16) + C(16x16 f32)
  return __builtin_amdgcn_wmma_f32_16x16x32_bf16(
      /*neg_a=*/false, a, /*neg_b=*/false, b,
      /*c_mod=*/(short)0, c, /*reuse_a=*/false, /*reuse_b=*/false);
}

// ============================================================
// 1) BN fold: scale = g*rsqrt(v+eps), shift = b - m*scale
//    Exactly 384 + 128 + 512 = 1024 channels -> one block.
// ============================================================
__global__ void __launch_bounds__(1024) prep_bn_kernel(
    const float* kvg, const float* kvb, const float* kvm, const float* kvv,
    const float* qg,  const float* qb,  const float* qm,  const float* qv,
    const float* pg,  const float* pb,  const float* pm,  const float* pv,
    float* sc_kv, float* sh_kv, float* sc_q, float* sh_q,
    float* sc_p,  float* sh_p) {
  const int t = threadIdx.x;
  if (t < 384) {
    float s = kvg[t] * rsqrtf(kvv[t] + 1e-5f);
    sc_kv[t] = s; sh_kv[t] = kvb[t] - kvm[t] * s;
  } else if (t < 512) {
    int i = t - 384;
    float s = qg[i] * rsqrtf(qv[i] + 1e-5f);
    sc_q[i] = s; sh_q[i] = qb[i] - qm[i] * s;
  } else {
    int i = t - 512;
    float s = pg[i] * rsqrtf(pv[i] + 1e-5f);
    sc_p[i] = s; sh_p[i] = pb[i] - pm[i] * s;
  }
}

// ============================================================
// 2) Convert weights to bf16 (kept [O][C] row-major so WMMA
//    B-fragments are two contiguous 8-element k-chunks).
// ============================================================
__global__ void __launch_bounds__(256) prep_w_kernel(
    const float* kvw, const float* qw, const float* pw,
    __bf16* Wkv, __bf16* Wq, __bf16* Wp) {
  int i = blockIdx.x * 256 + threadIdx.x;           // grid covers 262144
  if (i < 98304)        Wkv[i]          = (__bf16)kvw[i];
  else if (i < 131072)  Wq[i - 98304]   = (__bf16)qw[i - 98304];
  else if (i < 262144)  Wp[i - 131072]  = (__bf16)pw[i - 131072];
}

// ============================================================
// 3) Pack x (B,C,28,28) -> token-major bf16 Xt[b,hw,c] and the
//    stride-2 subsample Xq[b,n,c].
// ============================================================
__global__ void __launch_bounds__(256) pack_x_kernel(
    const float* __restrict__ x, __bf16* __restrict__ Xt,
    __bf16* __restrict__ Xq) {
  const int b = blockIdx.x;         // 0..63
  const int part = blockIdx.y;      // 0..3
  const int tid = threadIdx.x;
  const int hw = part * 196 + tid;  // valid when tid < 196
  if (tid >= 196) return;
  const int n = tid;
  const int src_q = (n / 14) * 56 + (n % 14) * 2;  // (2h)*28 + 2w
  for (int c = 0; c < CIN; ++c) {
    const float* xc = x + (b * CIN + c) * NKV;
    Xt[(b * NKV + hw) * CIN + c] = (__bf16)xc[hw];
    if (part == 0)
      Xq[(b * NQ + n) * CIN + c] = (__bf16)xc[src_q];
  }
}

// ---------- fragment loaders (ISA 7.12.2 layouts, wave32) ----------
// A 16x32 bf16 row-major: per-lane row = lane&15, k-chunks at
// {kb, kb+16}, kb = 8*(lane>=16).
__device__ __forceinline__ v16bf load_a(const __bf16* Arow, int kb) {
  union { v16bf v; v8bf h[2]; } u;
  u.h[0] = *(const v8bf*)(Arow + kb);
  u.h[1] = *(const v8bf*)(Arow + kb + 16);
  return u.v;
}
// B 32x16 bf16 from B^T storage [N][K]: k-chunks at {kb2/2, kb2/2+16},
// kb2 = 16*(lane>=16), col = lane&15.
__device__ __forceinline__ v16bf load_bT(const __bf16* Bcol, int kb2) {
  union { v16bf v; v8bf h[2]; } u;
  u.h[0] = *(const v8bf*)(Bcol + kb2 / 2);
  u.h[1] = *(const v8bf*)(Bcol + kb2 / 2 + 16);
  return u.v;
}

// ============================================================
// 4) KV GEMM: [50176 x 256] x [256 x 384] -> BN+hardswish ->
//    K[b,h,m,16] bf16 and V[b,h,32,m] bf16.
//    The 16x256 bf16 A tile (8KB) is DMA'd to LDS once per block
//    with the Tensor Data Mover (TDM), then all 8 waves feed
//    their WMMA A-fragments from LDS (ds_read_b128).
// ============================================================
__global__ void __launch_bounds__(256) gemm_kv_kernel(
    const __bf16* __restrict__ Xt, const __bf16* __restrict__ W,
    const float* __restrict__ sc, const float* __restrict__ sh,
    __bf16* __restrict__ Kb, __bf16* __restrict__ Vb) {
  __shared__ __bf16 sAt[16 * CIN];                  // 8 KB A tile
  const int lane = threadIdx.x & 31;
  const int wv = threadIdx.x >> 5;
  const int mt = blockIdx.x;                        // 0..3135
  const int kb = (lane & 16) ? 8 : 0;
  const int kb2 = (lane & 16) ? 16 : 0;
  const int col = lane & 15;

  if (wv == 0) {
    // Tensor DMA descriptor (D#), 2D tile: 16 rows x 256 cols, 2B elems.
    unsigned long long ga =
        (unsigned long long)(size_t)(Xt + (size_t)mt * 16 * CIN);
    unsigned lds = (unsigned)(size_t)(void*)sAt;    // LDS byte offset
    v4u g0;
    g0[0] = 1u;                                     // count=1 (user D#)
    g0[1] = lds;                                    // lds_addr [63:32]
    g0[2] = (unsigned)(ga & 0xffffffffu);           // global_addr lo
    g0[3] = (unsigned)((ga >> 32) & 0x01ffffffu)    // global_addr hi [120:96]
            | (2u << 30);                           // type=2 ("image") [127:126]
    v8u g1;
    g1[0] = 1u << 16;                               // data_size=1 (2B) [17:16]
    g1[1] = (unsigned)CIN << 16;                    // tensor_dim0 lo16 @bit48
    g1[2] = 16u << 16;                              // tensor_dim1 lo16 @bit80
    g1[3] = (unsigned)CIN << 16;                    // tile_dim0 [127:112]
    g1[4] = 16u;                                    // tile_dim1 [143:128]
    g1[5] = (unsigned)CIN;                          // tensor_dim0_stride lo32
    g1[6] = 0u;                                     // stride hi / dim1_stride lo
    g1[7] = 0u;
    asm volatile("tensor_load_to_lds %0, %1, null, null"
                 :: "s"(g0), "s"(g1) : "memory");
    __builtin_amdgcn_s_wait_tensorcnt(0);
  }
  __syncthreads();

  const __bf16* Ar = sAt + (lane & 15) * CIN;       // LDS A row

  for (int nt = wv; nt < 24; nt += 8) {
    const int o = nt * 16 + col;
    const __bf16* Bc = W + o * CIN;
    v8f acc = {};
#pragma unroll
    for (int k0 = 0; k0 < CIN; k0 += 32) {
      __builtin_prefetch(Bc + k0 + 64, 0, 0);       // global_prefetch_b8
      v16bf a = load_a(Ar + k0, kb);                // ds_read_b128 x2
      v16bf b = load_bT(Bc + k0, kb2);
      acc = wmma_bf16(a, b, acc);
    }
    const float s = sc[o], t = sh[o];
    const int h = o / 48, dd = o % 48;
#pragma unroll
    for (int r = 0; r < 8; ++r) {
      int tok = mt * 16 + ((lane & 16) ? r + 8 : r);
      int bb = tok / NKV, m = tok % NKV;
      float v = hswish(acc[r] * s + t);
      if (dd < KDIM)
        Kb[((bb * NHEAD + h) * NKV + m) * KDIM + dd] = (__bf16)v;
      else
        Vb[((bb * NHEAD + h) * DVDIM + (dd - KDIM)) * NKV + m] = (__bf16)v;
    }
  }
}

// ============================================================
// 5) Q GEMM: [12544 x 256] x [256 x 128] -> Q[b,h,n,16] bf16.
// ============================================================
__global__ void __launch_bounds__(256) gemm_q_kernel(
    const __bf16* __restrict__ Xq, const __bf16* __restrict__ W,
    const float* __restrict__ sc, const float* __restrict__ sh,
    __bf16* __restrict__ Qb) {
  const int lane = threadIdx.x & 31;
  const int wv = threadIdx.x >> 5;                  // nt = wv (8 tiles)
  const int mt = blockIdx.x;                        // 0..783
  const int rowA = mt * 16 + (lane & 15);
  const int kb = (lane & 16) ? 8 : 0;
  const int kb2 = (lane & 16) ? 16 : 0;
  const int col = lane & 15;
  const int o = wv * 16 + col;
  const __bf16* Ar = Xq + rowA * CIN;
  const __bf16* Bc = W + o * CIN;

  v8f acc = {};
#pragma unroll
  for (int k0 = 0; k0 < CIN; k0 += 32)
    acc = wmma_bf16(load_a(Ar + k0, kb), load_bT(Bc + k0, kb2), acc);

  const float s = sc[o], t = sh[o];
  const int h = o >> 4, d = o & 15;
#pragma unroll
  for (int r = 0; r < 8; ++r) {
    int tok = mt * 16 + ((lane & 16) ? r + 8 : r);
    int bb = tok / NQ, n = tok % NQ;
    Qb[((bb * NHEAD + h) * NQ + n) * KDIM + d] = (__bf16)hswish(acc[r] * s + t);
  }
}

// ============================================================
// 6) Attention: one block per (query-tile, b*h). 128 threads.
//    Phase1: logits (WMMA, K padded 16->32) + bias gather -> LDS bf16
//    Phase2: softmax over 784 (pad cols zeroed)
//    Phase3: attn x V^T (WMMA over K=800) -> hardswish -> OutTok bf16
// ============================================================
__global__ void __launch_bounds__(128) attn_kernel(
    const __bf16* __restrict__ Qb, const __bf16* __restrict__ Kb,
    const __bf16* __restrict__ Vb, const float* __restrict__ ab,
    const int* __restrict__ idxs, int n_off, __bf16* __restrict__ OutTok) {
  __shared__ __bf16 sA[16 * ATTN_LD];               // 25.6 KB
  __shared__ float red[16][8];
  __shared__ float stat[16];

  const int tid = threadIdx.x, lane = tid & 31, wv = tid >> 5;
  const int qt = blockIdx.x;                        // 0..12
  const int bh = blockIdx.y;                        // 0..511
  const int b = bh >> 3, h = bh & 7;
  const int col = lane & 15;
  const int hi = (lane & 16) ? 1 : 0;

  // zero the 784..799 pad columns
  for (int i = tid; i < 16 * 16; i += 128)
    sA[(i >> 4) * ATTN_LD + NKV + (i & 15)] = (__bf16)0.f;

  // ---- phase 1: logits = Q^T K * scale + bias ----
  union { v16bf v; v8bf h2[2]; } afr;
  afr.h2[1] = zero8();                              // k=16..31 pad (d<16)
  {
    const int nq = qt * 16 + col;                   // A row = query
    if (nq < NQ)
      afr.h2[0] = *(const v8bf*)(Qb + (bh * NQ + nq) * KDIM + hi * 8);
    else
      afr.h2[0] = zero8();
  }
  for (int kt = wv; kt < 49; kt += 4) {
    const int m0 = kt * 16;
    union { v16bf v; v8bf h2[2]; } bfr;
    {
      const __bf16* kp = Kb + (bh * NKV + m0 + col) * KDIM;
      bfr.h2[0] = *(const v8bf*)(kp + hi * 8);
      bfr.h2[1] = zero8();
    }
    v8f acc = {};
    acc = wmma_bf16(afr.v, bfr.v, acc);
#pragma unroll
    for (int r = 0; r < 8; ++r) {
      int qrow = hi ? r + 8 : r;
      int n = qt * 16 + qrow;
      int m = m0 + col;
      float lg = 0.f;
      if (n < NQ) lg = acc[r] * 0.25f + ab[h * n_off + idxs[n * NKV + m]];
      sA[qrow * ATTN_LD + m] = (__bf16)lg;
    }
  }
  __syncthreads();

  // ---- phase 2: softmax over m (rows: 8 threads x 98 cols) ----
  {
    const int q = tid >> 3, part = tid & 7;
    const int c0 = part * 98, c1 = c0 + 98;
    float mx = -1e30f;
    for (int c = c0; c < c1; ++c) mx = fmaxf(mx, (float)sA[q * ATTN_LD + c]);
    red[q][part] = mx;
    __syncthreads();
    if (part == 0) {
      float m2 = red[q][0];
#pragma unroll
      for (int i = 1; i < 8; ++i) m2 = fmaxf(m2, red[q][i]);
      stat[q] = m2;
    }
    __syncthreads();
    const float M = stat[q];
    float s = 0.f;
    for (int c = c0; c < c1; ++c) {
      float e = __expf((float)sA[q * ATTN_LD + c] - M);
      s += e;
      sA[q * ATTN_LD + c] = (__bf16)e;
    }
    red[q][part] = s;
    __syncthreads();
    if (part == 0) {
      float t2 = 0.f;
#pragma unroll
      for (int i = 0; i < 8; ++i) t2 += red[q][i];
      stat[q] = t2;
    }
    __syncthreads();
    const float inv = 1.f / stat[q];
    for (int c = c0; c < c1; ++c)
      sA[q * ATTN_LD + c] = (__bf16)((float)sA[q * ATTN_LD + c] * inv);
  }
  __syncthreads();

  // ---- phase 3: out[n][d] = sum_m attn[n][m] V[d][m] ----
  if (wv < 2) {
    const int dt = wv;                              // d-tile 0..1
    const int rowA = lane & 15;
    const int kb = hi ? 8 : 0;
    const int kb2 = hi ? 16 : 0;
    const int d = dt * 16 + col;
    const __bf16* vbase = Vb + (bh * DVDIM + d) * NKV;
    v8f acc = {};
    for (int kt = 0; kt < 25; ++kt) {
      const int k0 = kt * 32;
      union { v16bf v; v8bf h2[2]; } a2, b2;
      a2.h2[0] = *(const v8bf*)(sA + rowA * ATTN_LD + k0 + kb2);
      a2.h2[1] = *(const v8bf*)(sA + rowA * ATTN_LD + k0 + kb2 + 8);
      if (k0 + kb2 < NKV) {                         // 16-chunks align to 784
        b2.h2[0] = *(const v8bf*)(vbase + k0 + kb2);
        b2.h2[1] = *(const v8bf*)(vbase + k0 + kb2 + 8);
      } else {
        b2.h2[0] = zero8(); b2.h2[1] = zero8();     // attn pad cols are 0 too
      }
      acc = wmma_bf16(a2.v, b2.v, acc);
    }
#pragma unroll
    for (int r = 0; r < 8; ++r) {
      int qrow = hi ? r + 8 : r;
      int n = qt * 16 + qrow;
      if (n < NQ)
        OutTok[(b * NQ + n) * 256 + h * DVDIM + d] = (__bf16)hswish(acc[r]);
    }
  }
}

// ============================================================
// 7) Proj GEMM: [12544 x 256] x [256 x 512] -> BN+hardswish ->
//    d_out (B,512,14,14) fp32.
// ============================================================
__global__ void __launch_bounds__(256) gemm_proj_kernel(
    const __bf16* __restrict__ T, const __bf16* __restrict__ W,
    const float* __restrict__ sc, const float* __restrict__ sh,
    float* __restrict__ out) {
  const int lane = threadIdx.x & 31;
  const int wv = threadIdx.x >> 5;
  const int mt = blockIdx.x;                        // 0..783
  const int rowA = mt * 16 + (lane & 15);
  const int kb = (lane & 16) ? 8 : 0;
  const int kb2 = (lane & 16) ? 16 : 0;
  const int col = lane & 15;
  const __bf16* Ar = T + rowA * 256;

  for (int nt = wv; nt < 32; nt += 8) {
    const int o = nt * 16 + col;
    const __bf16* Bc = W + o * 256;
    v8f acc = {};
#pragma unroll
    for (int k0 = 0; k0 < 256; k0 += 32)
      acc = wmma_bf16(load_a(Ar + k0, kb), load_bT(Bc + k0, kb2), acc);
    const float s = sc[o], t = sh[o];
#pragma unroll
    for (int r = 0; r < 8; ++r) {
      int tok = mt * 16 + ((lane & 16) ? r + 8 : r);
      int bb = tok / NQ, n = tok % NQ;
      out[(bb * 512 + o) * NQ + n] = hswish(acc[r] * s + t);
    }
  }
}

// ============================================================
extern "C" void kernel_launch(void* const* d_in, const int* in_sizes, int n_in,
                              void* d_out, int out_size, void* d_ws, size_t ws_size,
                              hipStream_t stream) {
  const float* x    = (const float*)d_in[0];
  const float* kv_w = (const float*)d_in[1];
  const float* kv_g = (const float*)d_in[2];
  const float* kv_b = (const float*)d_in[3];
  const float* kv_m = (const float*)d_in[4];
  const float* kv_v = (const float*)d_in[5];
  const float* q_w  = (const float*)d_in[6];
  const float* q_g  = (const float*)d_in[7];
  const float* q_b  = (const float*)d_in[8];
  const float* q_m  = (const float*)d_in[9];
  const float* q_v  = (const float*)d_in[10];
  const float* p_w  = (const float*)d_in[11];
  const float* p_g  = (const float*)d_in[12];
  const float* p_b  = (const float*)d_in[13];
  const float* p_m  = (const float*)d_in[14];
  const float* p_v  = (const float*)d_in[15];
  const float* ab   = (const float*)d_in[16];
  const int*   idxs = (const int*)d_in[17];
  const int n_off = in_sizes[16] / NHEAD;

  // ---- workspace carve-up (256B aligned) ----
  char* p = (char*)d_ws;
  auto alloc = [&](size_t bytes) -> char* {
    char* r = p; p += (bytes + 255) & ~(size_t)255; return r;
  };
  __bf16* Wkv   = (__bf16*)alloc(384 * 256 * sizeof(__bf16));
  __bf16* Wq    = (__bf16*)alloc(128 * 256 * sizeof(__bf16));
  __bf16* Wp    = (__bf16*)alloc(512 * 256 * sizeof(__bf16));
  float*  sc_kv = (float*)alloc(384 * 4);
  float*  sh_kv = (float*)alloc(384 * 4);
  float*  sc_q  = (float*)alloc(128 * 4);
  float*  sh_q  = (float*)alloc(128 * 4);
  float*  sc_p  = (float*)alloc(512 * 4);
  float*  sh_p  = (float*)alloc(512 * 4);
  __bf16* Xt    = (__bf16*)alloc((size_t)BATCH * NKV * CIN * sizeof(__bf16));
  __bf16* Xq    = (__bf16*)alloc((size_t)BATCH * NQ * CIN * sizeof(__bf16));
  __bf16* Kb    = (__bf16*)alloc((size_t)BATCH * NHEAD * NKV * KDIM * sizeof(__bf16));
  __bf16* Vb    = (__bf16*)alloc((size_t)BATCH * NHEAD * DVDIM * NKV * sizeof(__bf16));
  __bf16* Qb    = (__bf16*)alloc((size_t)BATCH * NHEAD * NQ * KDIM * sizeof(__bf16));
  __bf16* OutTok= (__bf16*)alloc((size_t)BATCH * NQ * 256 * sizeof(__bf16));

  prep_bn_kernel<<<1, 1024, 0, stream>>>(kv_g, kv_b, kv_m, kv_v,
                                         q_g, q_b, q_m, q_v,
                                         p_g, p_b, p_m, p_v,
                                         sc_kv, sh_kv, sc_q, sh_q, sc_p, sh_p);
  prep_w_kernel<<<1024, 256, 0, stream>>>(kv_w, q_w, p_w, Wkv, Wq, Wp);
  pack_x_kernel<<<dim3(BATCH, 4), 256, 0, stream>>>(x, Xt, Xq);
  gemm_kv_kernel<<<(BATCH * NKV) / 16, 256, 0, stream>>>(Xt, Wkv, sc_kv, sh_kv, Kb, Vb);
  gemm_q_kernel<<<(BATCH * NQ) / 16, 256, 0, stream>>>(Xq, Wq, sc_q, sh_q, Qb);
  attn_kernel<<<dim3(13, BATCH * NHEAD), 128, 0, stream>>>(Qb, Kb, Vb, ab, idxs, n_off, OutTok);
  gemm_proj_kernel<<<(BATCH * NQ) / 16, 256, 0, stream>>>(OutTok, Wp, sc_p, sh_p, (float*)d_out);
}
